// BboxInteractionNetwork_84902913507459
// MI455X (gfx1250) — matrix operand order repaired and checked by reading, not compiled, not run on hardware
//
#include <hip/hip_runtime.h>
#include <stdint.h>

// ---------------- problem constants ----------------
#define E        128
#define NB       32
#define NT       512
#define NC       3
#define NTOT     (NB * NT * NC)     // 49152 scenes
#define NSAMP    16                 // scenes per block
#define THREADS  256                // 8 wave32
#define NBLOCKS  (NTOT / NSAMP)     // 3072

// ---------------- LDS layout (bytes) ----------------
#define OFF_W1    0                 // 5 x [128][128] bf16 transposed [n][k] = 163840
#define OFF_B1    163840            // 5 x [128] f32                         =   2560
#define OFF_W0    166400            // [16][128] f32                         =   8192
#define OFF_B0    174592            // 3 x [128] f32                         =   1536
#define OFF_XT    176128            // [16][16] f32                          =   1024
#define OFF_ACT0  177152            // [256][128] bf16                       =  65536
#define OFF_DYNF  242688            // [64][128] f32 (reused for f1)         =  32768
#define OFF_DYNB  275456            // [64][128] bf16                        =  16384
#define OFF_F0    291840            // [64][128] bf16                        =  16384
#define SMEM_TOT  308224

// workspace layout: bf16 transposed weights then f32 biases
#define WS_WBYTES (5 * 128 * 128 * 2)     // 163840
#define WS_BOFF   WS_WBYTES

typedef __bf16 bf16;
typedef bf16     v16bf __attribute__((ext_vector_type(16)));
typedef float    v8f   __attribute__((ext_vector_type(8)));
typedef uint32_t u32x8 __attribute__((ext_vector_type(8)));
typedef uint32_t u32x4 __attribute__((ext_vector_type(4)));

__device__ __forceinline__ uint16_t f2bf(float f) {
    uint32_t b = __builtin_bit_cast(uint32_t, f);
    uint32_t r = b + 0x7FFFu + ((b >> 16) & 1u);   // round-to-nearest-even
    return (uint16_t)(r >> 16);
}
__device__ __forceinline__ float bf2f(uint16_t h) {
    return __builtin_bit_cast(float, ((uint32_t)h) << 16);
}

__device__ __forceinline__ v8f wmma_bf16(u32x8 a, u32x8 b, v8f c) {
    return __builtin_amdgcn_wmma_f32_16x16x32_bf16(
        false, __builtin_bit_cast(v16bf, a),
        false, __builtin_bit_cast(v16bf, b),
        (short)0, c, false, false);
}

// A fragment (16x32 bf16, row-major LDS, 64 dwords/row): two 16B runs/lane.
__device__ __forceinline__ u32x8 load_a_frag(const uint32_t* base_u, int row0,
                                             int kc, int m15, int hi) {
    const u32x4* p = (const u32x4*)(base_u + (row0 + m15) * 64 + kc * 16);
    u32x4 lo = p[hi];
    u32x4 hi4 = p[2 + hi];
    return __builtin_shufflevector(lo, hi4, 0, 1, 2, 3, 4, 5, 6, 7);
}

// B fragment (32x16 bf16) from weight transposed [n][k]: two 16B runs/lane.
__device__ __forceinline__ u32x8 load_b_frag(const uint32_t* wT_u, int col0,
                                             int kc, int m15, int hi) {
    const u32x4* p = (const u32x4*)(wT_u + (col0 + m15) * 64 + kc * 16);
    u32x4 lo = p[hi];
    u32x4 hi4 = p[2 + hi];
    return __builtin_shufflevector(lo, hi4, 0, 1, 2, 3, 4, 5, 6, 7);
}

// Stage-1 epilogue: batched residual loads -> FMAs -> non-returning LDS atomics.
// rowoff[r] = 128 * dyn_row precomputed per (R,hi).
__device__ __forceinline__ void stage1_epilogue(v8f c, int R, int col, int hi,
                                                const int* rowoff,
                                                const float* bia,
                                                const uint16_t* act0,
                                                float* dynf) {
    float resid[8];
#pragma unroll
    for (int r = 0; r < 8; ++r)
        resid[r] = bf2f(act0[(R + r + hi * 8) * 128 + col]);
    float bv = bia[col];
#pragma unroll
    for (int r = 0; r < 8; ++r)
        atomicAdd(&dynf[rowoff[r] + col], c[r] + bv + resid[r]);
}

// ---------------- prep kernel: weights -> bf16 transposed in workspace -------
__global__ __launch_bounds__(256)
void prep_weights(const float* __restrict__ sa1_w, const float* __restrict__ sb1_w,
                  const float* __restrict__ rel1_w, const float* __restrict__ aff0_w,
                  const float* __restrict__ aff1_w,
                  const float* __restrict__ sa1_b, const float* __restrict__ sb1_b,
                  const float* __restrict__ rel1_b, const float* __restrict__ aff0_b,
                  const float* __restrict__ aff1_b,
                  uint16_t* __restrict__ wsT, float* __restrict__ wsB)
{
    int gid = blockIdx.x * 256 + threadIdx.x;       // 81920 threads total
    const float* W[5] = {sa1_w, sb1_w, rel1_w, aff0_w, aff1_w};
    int m = gid >> 14;
    int r = gid & 16383;
    int k = r >> 7, n = r & 127;
    wsT[m * 16384 + n * 128 + k] = f2bf(W[m][k * 128 + n]);
    if (gid < 640) {
        const float* Bb[5] = {sa1_b, sb1_b, rel1_b, aff0_b, aff1_b};
        wsB[gid] = Bb[gid >> 7][gid & 127];
    }
}

__global__ __launch_bounds__(THREADS)
void bbox_in_fused(const float* __restrict__ x,
                   const float* __restrict__ sa0_w, const float* __restrict__ sa0_b,
                   const float* __restrict__ sb0_w, const float* __restrict__ sb0_b,
                   const float* __restrict__ rel0_w, const float* __restrict__ rel0_b,
                   const uint16_t* __restrict__ wsT, const float* __restrict__ wsB,
                   float* __restrict__ out_xf, float* __restrict__ out_fin)
{
    extern __shared__ char smem[];
    uint16_t* w1   = (uint16_t*)(smem + OFF_W1);
    float*    b1   = (float*)   (smem + OFF_B1);
    float*    w0   = (float*)   (smem + OFF_W0);
    float*    b0   = (float*)   (smem + OFF_B0);
    float*    xt   = (float*)   (smem + OFF_XT);
    uint16_t* act0 = (uint16_t*)(smem + OFF_ACT0);
    float*    dynf = (float*)   (smem + OFF_DYNF);
    uint16_t* dynb = (uint16_t*)(smem + OFF_DYNB);
    uint16_t* f0b  = (uint16_t*)(smem + OFF_F0);

    const int tid  = threadIdx.x;
    const int blk  = blockIdx.x;
    const int lane = tid & 31;
    const int wave = tid >> 5;
    const int m15  = lane & 15;
    const int hi   = lane >> 4;

    // -------- async DMA of prepped bf16 weights into LDS (overlaps stage 0) ---
    // 163840 B total; 8 waves x 40 iters x 32 lanes x 16 B. Tracked by ASYNCcnt.
    {
        uint32_t base_off = (uint32_t)(wave * 20480 + lane * 16);
#pragma unroll 4
        for (int it = 0; it < 40; ++it) {
            uint32_t goff = base_off + (uint32_t)(it * 512);
            uint32_t lds_addr = (uint32_t)OFF_W1 + goff;
            asm volatile("global_load_async_to_lds_b128 %0, %1, %2"
                         :: "v"(lds_addr), "v"(goff), "s"(wsT)
                         : "memory");
        }
    }

    // -------- small staging (normal loads): biases, first-layer weights, x ----
    for (int i = tid; i < 640; i += THREADS) b1[i] = wsB[i];
    for (int i = tid; i < 4 * 128; i += THREADS) w0[i]            = sa0_w[i];
    for (int i = tid; i < 4 * 128; i += THREADS) w0[4 * 128 + i]  = sb0_w[i];
    for (int i = tid; i < 8 * 128; i += THREADS) w0[8 * 128 + i]  = rel0_w[i];
    for (int i = tid; i < 128; i += THREADS) {
        b0[i] = sa0_b[i]; b0[128 + i] = sb0_b[i]; b0[256 + i] = rel0_b[i];
    }
    {
        float v = x[blk * 256 + tid];
        xt[tid] = v;
        out_xf[blk * 256 + tid] = v;           // xf passthrough output
    }
    for (int i = tid; i < 64 * 128; i += THREADS) dynf[i] = 0.f;
    __syncthreads();

    // ---------------- stage 0: first-layer MLPs (K=4/8) -> act0 bf16 ----------
    for (int idx = tid; idx < 256 * 128; idx += THREADS) {
        int row = idx >> 7, col = idx & 127;
        float acc;
        if (row < 16) {
            const float* in = xt + row * 16;                 // agent = xf[:,0,:]
            acc = b0[col];
#pragma unroll
            for (int k = 0; k < 4; ++k) acc = fmaf(in[k], w0[k * 128 + col], acc);
        } else if (row < 64) {
            int i2 = row - 16; int s = i2 / 3; int bx = i2 % 3;
            const float* in = xt + s * 16 + 4 * (bx + 1);    // xf[:,1+bx,:]
            acc = b0[128 + col];
#pragma unroll
            for (int k = 0; k < 4; ++k) acc = fmaf(in[k], w0[(4 + k) * 128 + col], acc);
        } else {
            int i2 = row - 64; int s = i2 / 12; int p = i2 % 12;
            int pi = p / 3; int pj = p % 3; pj += (pj >= pi); // _PAIRS (i-major)
            const float* ia = xt + s * 16 + 4 * pi;
            const float* ib = xt + s * 16 + 4 * pj;
            acc = b0[256 + col];
#pragma unroll
            for (int k = 0; k < 4; ++k) acc = fmaf(ia[k], w0[(8 + k)  * 128 + col], acc);
#pragma unroll
            for (int k = 0; k < 4; ++k) acc = fmaf(ib[k], w0[(12 + k) * 128 + col], acc);
        }
        act0[idx] = f2bf(fmaxf(acc, 0.f));
    }
    // weights must be resident before stage 1 consumes them
    asm volatile("s_wait_asynccnt 0" ::: "memory");
    __syncthreads();

    const uint32_t* act0_u = (const uint32_t*)act0;
    const uint32_t* w1_u   = (const uint32_t*)w1;

    // ---------------- stage 1: second-layer GEMMs + scatter into dyn ----------
    // 256x128 output = 16 row-tiles x 8 col-tiles; wave owns rt {2w, 2w+1}.
    // Col-tiles in pairs with independent accumulators (hide WMMA RAW hazard).
#pragma unroll
    for (int rti = 0; rti < 2; ++rti) {
        int rt = wave * 2 + rti;
        int R  = rt * 16;
        int wsel = (rt == 0) ? 0 : (rt < 4 ? 1 : 2);         // sa1 / sb1 / rel1
        const uint32_t* wT  = w1_u + wsel * (128 * 64);
        const float*    bia = b1   + wsel * 128;

        int rowoff[8];                                        // dyn row offsets
#pragma unroll
        for (int r = 0; r < 8; ++r) {
            int g = R + r + hi * 8;
            int s, slot;
            if (g < 16)      { s = g;                 slot = 0; }
            else if (g < 64) { int i2 = g - 16; s = i2 / 3;  slot = i2 % 3 + 1; }
            else             { int i2 = g - 64; s = i2 / 12; slot = (i2 % 12) / 3; }
            rowoff[r] = (s * 4 + slot) * 128;
        }

        u32x8 afr[4];
#pragma unroll
        for (int kc = 0; kc < 4; ++kc) afr[kc] = load_a_frag(act0_u, R, kc, m15, hi);

#pragma unroll
        for (int ct2 = 0; ct2 < 4; ++ct2) {
            int ctA = ct2 * 2, ctB = ct2 * 2 + 1;
            v8f c0 = {}, c1 = {};
#pragma unroll
            for (int kc = 0; kc < 4; ++kc) {
                u32x8 bA = load_b_frag(wT, ctA * 16, kc, m15, hi);
                u32x8 bB = load_b_frag(wT, ctB * 16, kc, m15, hi);
                c0 = wmma_bf16(afr[kc], bA, c0);
                c1 = wmma_bf16(afr[kc], bB, c1);
            }
            stage1_epilogue(c0, R, ctA * 16 + m15, hi, rowoff, bia, act0, dynf);
            stage1_epilogue(c1, R, ctB * 16 + m15, hi, rowoff, bia, act0, dynf);
        }
    }
    __syncthreads();

    for (int i = tid; i < 64 * 128; i += THREADS) dynb[i] = f2bf(dynf[i]);
    __syncthreads();

    // ---------------- stage 2: f0 = relu(dyn @ aff0 + b) ----------------------
    const uint32_t* dynb_u = (const uint32_t*)dynb;
    {
        int rt = wave >> 1;
        int R  = rt * 16;
        const uint32_t* wT  = w1_u + 3 * (128 * 64);
        const float*    bia = b1   + 3 * 128;
        u32x8 afr[4];
#pragma unroll
        for (int kc = 0; kc < 4; ++kc) afr[kc] = load_a_frag(dynb_u, R, kc, m15, hi);

#pragma unroll
        for (int ci = 0; ci < 2; ++ci) {
            int ctA = (wave & 1) * 4 + ci * 2, ctB = ctA + 1;
            v8f c0 = {}, c1 = {};
#pragma unroll
            for (int kc = 0; kc < 4; ++kc) {
                u32x8 bA = load_b_frag(wT, ctA * 16, kc, m15, hi);
                u32x8 bB = load_b_frag(wT, ctB * 16, kc, m15, hi);
                c0 = wmma_bf16(afr[kc], bA, c0);
                c1 = wmma_bf16(afr[kc], bB, c1);
            }
            int colA = ctA * 16 + m15, colB = ctB * 16 + m15;
            float bvA = bia[colA], bvB = bia[colB];
#pragma unroll
            for (int r = 0; r < 8; ++r) {
                int g = R + r + hi * 8;
                f0b[g * 128 + colA] = f2bf(fmaxf(c0[r] + bvA, 0.f));
                f0b[g * 128 + colB] = f2bf(fmaxf(c1[r] + bvB, 0.f));
            }
        }
    }
    __syncthreads();

    // ---------------- stage 3: f1 = f0 @ aff1 + b + f0 -> dynf (reuse) --------
    const uint32_t* f0_u = (const uint32_t*)f0b;
    {
        int rt = wave >> 1;
        int R  = rt * 16;
        const uint32_t* wT  = w1_u + 4 * (128 * 64);
        const float*    bia = b1   + 4 * 128;
        u32x8 afr[4];
#pragma unroll
        for (int kc = 0; kc < 4; ++kc) afr[kc] = load_a_frag(f0_u, R, kc, m15, hi);

#pragma unroll
        for (int ci = 0; ci < 2; ++ci) {
            int ctA = (wave & 1) * 4 + ci * 2, ctB = ctA + 1;
            v8f c0 = {}, c1 = {};
#pragma unroll
            for (int kc = 0; kc < 4; ++kc) {
                u32x8 bA = load_b_frag(wT, ctA * 16, kc, m15, hi);
                u32x8 bB = load_b_frag(wT, ctB * 16, kc, m15, hi);
                c0 = wmma_bf16(afr[kc], bA, c0);
                c1 = wmma_bf16(afr[kc], bB, c1);
            }
            int colA = ctA * 16 + m15, colB = ctB * 16 + m15;
            float residA[8], residB[8];
#pragma unroll
            for (int r = 0; r < 8; ++r) {
                int g = R + r + hi * 8;
                residA[r] = bf2f(f0b[g * 128 + colA]);
                residB[r] = bf2f(f0b[g * 128 + colB]);
            }
            float bvA = bia[colA], bvB = bia[colB];
#pragma unroll
            for (int r = 0; r < 8; ++r) {
                int g = R + r + hi * 8;
                dynf[g * 128 + colA] = c0[r] + bvA + residA[r];
                dynf[g * 128 + colB] = c1[r] + bvB + residB[r];
            }
        }
    }
    __syncthreads();

    // ---------------- final: mean over 4 slots -> out_fin ---------------------
    for (int idx = tid; idx < NSAMP * 128; idx += THREADS) {
        int s = idx >> 7, col = idx & 127;
        float m = 0.25f * (dynf[(s * 4 + 0) * 128 + col] +
                           dynf[(s * 4 + 1) * 128 + col] +
                           dynf[(s * 4 + 2) * 128 + col] +
                           dynf[(s * 4 + 3) * 128 + col]);
        int n  = blk * NSAMP + s;
        int cc = n % 3, bt = n / 3;              // final laid out (B*T, 3*128)
        out_fin[bt * 384 + cc * 128 + col] = m;
    }
}

extern "C" void kernel_launch(void* const* d_in, const int* in_sizes, int n_in,
                              void* d_out, int out_size, void* d_ws, size_t ws_size,
                              hipStream_t stream) {
    const float* x      = (const float*)d_in[0];
    const float* sa0_w  = (const float*)d_in[1];
    const float* sa0_b  = (const float*)d_in[2];
    const float* sa1_w  = (const float*)d_in[3];
    const float* sa1_b  = (const float*)d_in[4];
    const float* sb0_w  = (const float*)d_in[5];
    const float* sb0_b  = (const float*)d_in[6];
    const float* sb1_w  = (const float*)d_in[7];
    const float* sb1_b  = (const float*)d_in[8];
    const float* rel0_w = (const float*)d_in[9];
    const float* rel0_b = (const float*)d_in[10];
    const float* rel1_w = (const float*)d_in[11];
    const float* rel1_b = (const float*)d_in[12];
    const float* aff0_w = (const float*)d_in[13];
    const float* aff0_b = (const float*)d_in[14];
    const float* aff1_w = (const float*)d_in[15];
    const float* aff1_b = (const float*)d_in[16];

    float* out_xf  = (float*)d_out;                 // xf: N*16 floats
    float* out_fin = out_xf + (size_t)NTOT * 16;    // final: B*T*384 floats

    uint16_t* wsT = (uint16_t*)d_ws;
    float*    wsB = (float*)((char*)d_ws + WS_BOFF);

    prep_weights<<<320, 256, 0, stream>>>(sa1_w, sb1_w, rel1_w, aff0_w, aff1_w,
                                          sa1_b, sb1_b, rel1_b, aff0_b, aff1_b,
                                          wsT, wsB);

    (void)hipFuncSetAttribute((const void*)bbox_in_fused,
                              hipFuncAttributeMaxDynamicSharedMemorySize,
                              (int)SMEM_TOT);

    bbox_in_fused<<<NBLOCKS, THREADS, SMEM_TOT, stream>>>(
        x, sa0_w, sa0_b, sb0_w, sb0_b, rel0_w, rel0_b,
        wsT, wsB, out_xf, out_fin);
}
